// OrthogonalGDIFNet_67190468379186
// MI455X (gfx1250) — matrix operand order repaired and checked by reading, not compiled, not run on hardware
//
#include <hip/hip_runtime.h>
#include <hip/hip_bf16.h>

typedef __attribute__((ext_vector_type(16))) _Float16 v16h;
typedef __attribute__((ext_vector_type(8)))  _Float16 v8h;
typedef __attribute__((ext_vector_type(8)))  float    v8f;

#define NN    50000
#define EE    800000
#define ETOT  (EE + NN)
#define ROWT  (NN / 16)   // 3125 exact -> no row guards, EXEC all-1s for WMMA

__device__ __forceinline__ float gelu_f(float x) {
  return 0.5f * x * (1.0f + erff(x * 0.70710678118654752f));  // exact erf GELU
}
__device__ __forceinline__ unsigned ford(float f) {   // order-preserving f32->u32
  unsigned u = __float_as_uint(f);
  return (u >> 31) ? ~u : (u | 0x80000000u);
}
__device__ __forceinline__ float fdec(unsigned s) {
  return (s >> 31) ? __uint_as_float(s ^ 0x80000000u) : __uint_as_float(~s);
}

// ---------------------------------------------------------------- weights -> f16 (padded / fused)
struct WPrep {
  const float *w1f, *w1fi, *w1c, *w1h;   // enc_w1_* : [in_r,64]
  const float *w2;                       // enc_w2   : [4,64,64]
  const float *wl, *wr;                  // gat_wl/wr: [64,256]
  const float *hw1;                      // head_w1  : [4,64,32]
  _Float16 *W1pad, *W2c, *WLR, *HW1c;
};

__global__ void k_prep_weights(WPrep p) {
  int i = blockIdx.x * blockDim.x + threadIdx.x;  // 65536 threads total
  if (i < 8192) {                                  // W1pad: 4 x [32,64], zero-pad K
    int r = i >> 11, rem = i & 2047, k = rem >> 6, c = rem & 63;
    const int widths[4] = {20, 18, 16, 18};
    const float* srcs[4] = {p.w1f, p.w1fi, p.w1c, p.w1h};
    p.W1pad[i] = (_Float16)((k < widths[r]) ? srcs[r][k * 64 + c] : 0.0f);
  } else if (i < 24576) {                          // W2c: 4 x [64,64]
    int j = i - 8192;  p.W2c[j] = (_Float16)p.w2[j];
  } else if (i < 57344) {                          // WLR: [64,512] = wl | wr
    int j = i - 24576; int k = j >> 9, c = j & 511;
    p.WLR[j] = (_Float16)((c < 256) ? p.wl[k * 256 + c] : p.wr[k * 256 + c - 256]);
  } else if (i < 65536) {                          // HW1c: 4 x [64,32]
    int j = i - 57344; p.HW1c[j] = (_Float16)p.hw1[j];
  }
}

// ---------------------------------------------------------------- concat modalities -> f16 [N,32]
__global__ void k_stage_x(const float* __restrict__ xa, int da,
                          const float* __restrict__ xb, int db,
                          _Float16* __restrict__ A1) {
  int i = blockIdx.x * blockDim.x + threadIdx.x;
  if (i >= NN * 32) return;
  int n = i >> 5, k = i & 31;
  float v = 0.0f;
  if (k < da)           v = xa[(size_t)n * da + k];
  else if (k < da + db) v = xb[(size_t)n * db + (k - da)];
  A1[i] = (_Float16)v;
}

// ---------------------------------------------------------------- WMMA GEMM: out = act(A[N,K]@B[K,Ncols]+bias)
__global__ __launch_bounds__(32)
void k_wmma_gemm(const _Float16* __restrict__ A, const _Float16* __restrict__ B,
                 const float* __restrict__ bias, float* __restrict__ outF32,
                 _Float16* __restrict__ outF16, int K, int Ncols, int act) {
  const int lane = threadIdx.x;
  const int rowTile = blockIdx.x, colTile = blockIdx.y;
  const int row = lane & 15;
  const int kbase = (lane >> 4) << 3;      // ISA 16-bit A 16x32 layout
  v8f c = {0.f, 0.f, 0.f, 0.f, 0.f, 0.f, 0.f, 0.f};
  for (int kk = 0; kk < K; kk += 32) {
    const _Float16* ap = A + (size_t)(rowTile * 16 + row) * K + kk + kbase;
    v8h alo = *(const v8h*)ap;             // K = kbase + 0..7
    v8h ahi = *(const v8h*)(ap + 16);      // K = kbase + 16..23
    v16h a;
#pragma unroll
    for (int t = 0; t < 8; ++t) { a[t] = alo[t]; a[t + 8] = ahi[t]; }
    // B 32x16: lane holds column lane&15, K = kbase + 0..7 and kbase + 16..23
    v16h b;
#pragma unroll
    for (int t = 0; t < 8; ++t) {
      b[t] = B[(size_t)(kk + kbase + t) * Ncols + colTile * 16 + row];
      b[t + 8] = B[(size_t)(kk + kbase + 16 + t) * Ncols + colTile * 16 + row];
    }
    c = __builtin_amdgcn_wmma_f32_16x16x32_f16(false, a, false, b, (short)0, c,
                                               false, false);
  }
  const int col = colTile * 16 + (lane & 15);
  const int rbase = rowTile * 16 + ((lane >> 4) << 3);  // C: hi half-wave = rows+8
  const float bb = bias ? bias[col] : 0.0f;
#pragma unroll
  for (int v = 0; v < 8; ++v) {
    float val = c[v] + bb;
    if (act == 1) val = gelu_f(val);
    size_t o = (size_t)(rbase + v) * Ncols + col;
    if (outF32) outF32[o] = val;
    if (outF16) outF16[o] = (_Float16)val;
  }
}

// ---------------------------------------------------------------- init segment buffers
__global__ void k_init_seg(unsigned* __restrict__ M, float* __restrict__ D,
                           float* __restrict__ Gout) {
  int i = blockIdx.x * blockDim.x + threadIdx.x;
  if (i >= NN * 64) return;
  Gout[i] = 0.0f;
  if (i < NN * 4) { M[i] = 0x007FFFFFu; D[i] = 0.0f; }  // ford(-inf)
}

// ---------------------------------------------------------------- edge pass 1: logits + segment max
__global__ void k_edge_logits(const int* __restrict__ ei, const float* __restrict__ XLR,
                              const float* __restrict__ att, float* __restrict__ Ebuf,
                              unsigned* __restrict__ M) {
  int t = blockIdx.x * blockDim.x + threadIdx.x;
  if (t >= ETOT) return;
  int s, d;
  if (t < EE) { s = ei[t]; d = ei[EE + t]; } else { s = d = t - EE; }
  const float4* xl = (const float4*)(XLR + (size_t)s * 512);
  const float4* xr = (const float4*)(XLR + (size_t)d * 512 + 256);
  const float4* at = (const float4*)att;
#pragma unroll
  for (int h = 0; h < 4; ++h) {
    float acc = 0.0f;
#pragma unroll
    for (int q = 0; q < 16; ++q) {
      float4 a = xl[h * 16 + q], b = xr[h * 16 + q], w = at[h * 16 + q];
      float v;
      v = a.x + b.x; v = v > 0.f ? v : 0.2f * v; acc += v * w.x;
      v = a.y + b.y; v = v > 0.f ? v : 0.2f * v; acc += v * w.y;
      v = a.z + b.z; v = v > 0.f ? v : 0.2f * v; acc += v * w.z;
      v = a.w + b.w; v = v > 0.f ? v : 0.2f * v; acc += v * w.w;
    }
    Ebuf[(size_t)t * 4 + h] = acc;
    atomicMax(&M[d * 4 + h], ford(acc));
  }
}

// ---------------------------------------------------------------- edge pass 2: exp + segment sum
__global__ void k_edge_exp(const int* __restrict__ ei, float* __restrict__ Ebuf,
                           const unsigned* __restrict__ M, float* __restrict__ D) {
  int i = blockIdx.x * blockDim.x + threadIdx.x;
  if (i >= ETOT * 4) return;
  int t = i >> 2, h = i & 3;
  int d = (t < EE) ? ei[EE + t] : (t - EE);
  float ex = __expf(Ebuf[i] - fdec(M[d * 4 + h]));
  Ebuf[i] = ex;
  atomicAdd(&D[d * 4 + h], ex);
}

// ---------------------------------------------------------------- edge pass 3: aggregate, head-mean folded in
__global__ void k_edge_aggr(const int* __restrict__ ei, const float* __restrict__ Ebuf,
                            const float* __restrict__ D, const float* __restrict__ XLR,
                            float* __restrict__ Gout) {
  long long i = (long long)blockIdx.x * blockDim.x + threadIdx.x;
  if (i >= (long long)ETOT * 64) return;
  int t = (int)(i >> 6), c = (int)(i & 63);
  int s, d;
  if (t < EE) { s = ei[t]; d = ei[EE + t]; } else { s = d = t - EE; }
  float acc = 0.0f;
#pragma unroll
  for (int h = 0; h < 4; ++h)
    acc += (Ebuf[(size_t)t * 4 + h] / D[d * 4 + h]) * XLR[(size_t)s * 512 + h * 64 + c];
  atomicAdd(&Gout[(size_t)d * 64 + c], 0.25f * acc);
}

// ---------------------------------------------------------------- gelu(gat + bias) -> f16
__global__ void k_gat_finish(const float* __restrict__ Gout, const float* __restrict__ bias,
                             _Float16* __restrict__ Gf16) {
  int i = blockIdx.x * blockDim.x + threadIdx.x;
  if (i >= NN * 64) return;
  Gf16[i] = (_Float16)gelu_f(Gout[i] + bias[i & 63]);
}

// ---------------------------------------------------------------- final dot(32) + sigmoid
__global__ void k_head_out(const float* __restrict__ H1, const float* __restrict__ w2,
                           const float* __restrict__ b2, int r, float* __restrict__ out) {
  int n = blockIdx.x * blockDim.x + threadIdx.x;
  if (n >= NN) return;
  const float4* hp = (const float4*)(H1 + (size_t)n * 32);
  const float4* wp = (const float4*)w2;
  float acc = b2[r];
#pragma unroll
  for (int q = 0; q < 8; ++q) {
    float4 a = hp[q], w = wp[q];
    acc += a.x * w.x + a.y * w.y + a.z * w.z + a.w * w.w;
  }
  out[(size_t)n * 4 + r] = 1.0f / (1.0f + __expf(-acc));
}

// ================================================================ launcher
extern "C" void kernel_launch(void* const* d_in, const int* in_sizes, int n_in,
                              void* d_out, int out_size, void* d_ws, size_t ws_size,
                              hipStream_t stream) {
  const float* mod_a = (const float*)d_in[0];
  const float* mod_b = (const float*)d_in[1];
  const float* mod_c = (const float*)d_in[2];
  const float* mod_d = (const float*)d_in[3];
  const int*   ei    = (const int*)  d_in[4];
  const float* w1f   = (const float*)d_in[5];
  const float* w1fi  = (const float*)d_in[6];
  const float* w1c   = (const float*)d_in[7];
  const float* w1h   = (const float*)d_in[8];
  const float* eb1   = (const float*)d_in[9];
  const float* ew2   = (const float*)d_in[10];
  const float* eb2   = (const float*)d_in[11];
  const float* gwl   = (const float*)d_in[12];
  const float* gwr   = (const float*)d_in[13];
  const float* gatt  = (const float*)d_in[14];
  const float* gbias = (const float*)d_in[15];
  const float* hw1   = (const float*)d_in[16];
  const float* hb1   = (const float*)d_in[17];
  const float* hw2   = (const float*)d_in[18];
  const float* hb2   = (const float*)d_in[19];
  float* out = (float*)d_out;

  char* p = (char*)d_ws;
  auto alloc = [&](size_t bytes) -> void* {
    void* r = (void*)p; p += (bytes + 255) & ~(size_t)255; return r;
  };
  _Float16* W1pad = (_Float16*)alloc(4 * 32 * 64 * sizeof(_Float16));
  _Float16* W2c   = (_Float16*)alloc(4 * 64 * 64 * sizeof(_Float16));
  _Float16* WLR   = (_Float16*)alloc(64 * 512 * sizeof(_Float16));
  _Float16* HW1c  = (_Float16*)alloc(4 * 64 * 32 * sizeof(_Float16));
  _Float16* A1    = (_Float16*)alloc((size_t)NN * 32 * sizeof(_Float16));
  _Float16* H1h   = (_Float16*)alloc((size_t)NN * 64 * sizeof(_Float16));
  _Float16* Zh    = (_Float16*)alloc((size_t)NN * 64 * sizeof(_Float16));
  _Float16* Gh    = (_Float16*)alloc((size_t)NN * 64 * sizeof(_Float16));
  float*    XLR   = (float*)   alloc((size_t)NN * 512 * sizeof(float));
  float*    Ebuf  = (float*)   alloc((size_t)ETOT * 4 * sizeof(float));
  unsigned* Mbuf  = (unsigned*)alloc((size_t)NN * 4 * sizeof(unsigned));
  float*    Dbuf  = (float*)   alloc((size_t)NN * 4 * sizeof(float));
  float*    Gout  = (float*)   alloc((size_t)NN * 64 * sizeof(float));
  float*    H1f   = (float*)   alloc((size_t)NN * 32 * sizeof(float));

  WPrep wp{w1f, w1fi, w1c, w1h, ew2, gwl, gwr, hw1, W1pad, W2c, WLR, HW1c};
  k_prep_weights<<<65536 / 256, 256, 0, stream>>>(wp);

  const float* xsa[4] = {mod_a, mod_b, mod_c, mod_a};
  const int    dsa[4] = {8, 12, 6, 8};
  const float* xsb[4] = {mod_b, mod_c, mod_d, mod_d};
  const int    dsb[4] = {12, 6, 10, 10};
  const int TB = 256;

  for (int r = 0; r < 4; ++r) {
    float* latents_r = out + (size_t)4 * NN + (size_t)r * NN * 64;

    k_stage_x<<<(NN * 32 + TB - 1) / TB, TB, 0, stream>>>(xsa[r], dsa[r], xsb[r], dsb[r], A1);
    k_wmma_gemm<<<dim3(ROWT, 4), 32, 0, stream>>>(A1, W1pad + (size_t)r * 2048,
        eb1 + r * 64, nullptr, H1h, 32, 64, 1);
    k_wmma_gemm<<<dim3(ROWT, 4), 32, 0, stream>>>(H1h, W2c + (size_t)r * 4096,
        eb2 + r * 64, latents_r, Zh, 64, 64, 0);
    k_wmma_gemm<<<dim3(ROWT, 32), 32, 0, stream>>>(Zh, WLR, nullptr, XLR, nullptr, 64, 512, 0);

    k_init_seg<<<(NN * 64 + TB - 1) / TB, TB, 0, stream>>>(Mbuf, Dbuf, Gout);
    k_edge_logits<<<(ETOT + TB - 1) / TB, TB, 0, stream>>>(ei, XLR, gatt, Ebuf, Mbuf);
    k_edge_exp<<<(ETOT * 4 + TB - 1) / TB, TB, 0, stream>>>(ei, Ebuf, Mbuf, Dbuf);
    long long tot = (long long)ETOT * 64;
    k_edge_aggr<<<(int)((tot + TB - 1) / TB), TB, 0, stream>>>(ei, Ebuf, Dbuf, XLR, Gout);
    k_gat_finish<<<(NN * 64 + TB - 1) / TB, TB, 0, stream>>>(Gout, gbias, Gh);

    k_wmma_gemm<<<dim3(ROWT, 2), 32, 0, stream>>>(Gh, HW1c + (size_t)r * 2048,
        hb1 + r * 32, H1f, nullptr, 64, 32, 1);
    k_head_out<<<(NN + TB - 1) / TB, TB, 0, stream>>>(H1f, hw2 + r * 32, hb2, r, out);
  }
  (void)in_sizes; (void)n_in; (void)out_size; (void)ws_size;
}
